// BasicNCAModel_8675833937932
// MI455X (gfx1250) — matrix operand order repaired and checked by reading, not compiled, not run on hardware
//
#include <hip/hip_runtime.h>
#include <hip/hip_bf16.h>
#include <stdint.h>

typedef __attribute__((ext_vector_type(16))) _Float16 v16h;
typedef __attribute__((ext_vector_type(8)))  float    v8f;

#define CCH   48
#define HID   256
#define KPAD  160      // perception K padded 144 -> 160 (5 k-tiles of 32)
#define PSTR  168      // LDS row stride (halves) for perception (bank-safe)
#define HSTR  264      // LDS row stride (halves) for hidden (bank-safe)
#define PIX   64       // pixels per block (2 waves x 2 M-tiles)
#define XCOLS 66       // PIX + 2 halo columns
#define XSTR  49       // padded channel stride (floats): gcd(49,64)=1 -> no conflicts
#define TPB   64       // 2 waves

union Frag16 { v16h v; uint32_t d[8]; float4 f4[2]; };

__device__ __forceinline__ uint32_t hashu(uint32_t x) {
  x ^= x >> 16; x *= 0x7feb352du;
  x ^= x >> 15; x *= 0x846ca68bu;
  x ^= x >> 16; return x;
}

// A-fragment (16x32 f16) per ISA 7.12.2: lanes 0-15 = rows, K {0..7,16..23};
// lanes 16-31 same rows, K {8..15,24..31}. All addresses 16B aligned -> b128.
__device__ __forceinline__ v16h load_afrag(const _Float16* row, int kbase, int khi) {
  Frag16 u;
#pragma unroll
  for (int j = 0; j < 4; ++j)
    u.d[j] = *(const uint32_t*)(row + kbase + khi + 2 * j);
#pragma unroll
  for (int j = 0; j < 4; ++j)
    u.d[4 + j] = *(const uint32_t*)(row + kbase + khi + 16 + 2 * j);
  return u.v;
}

// ---- one-time weight swizzles into per-lane B-fragment order (f16) ----
// B frag: lane n (0..15) holds K kbase+{0..7,16..23}, lane n+16 holds K kbase+{8..15,24..31},
// 16 consecutive halves per lane -> two contiguous b128 loads per lane.
__global__ void prep_w1(const float* __restrict__ w1, _Float16* __restrict__ out) {
  int idx = blockIdx.x * blockDim.x + threadIdx.x;
  if (idx >= 16 * 5 * 512) return;
  int j = idx & 15, lane = (idx >> 4) & 31, frag = idx >> 9;
  int kt = frag % 5, nt = frag / 5;
  int n = nt * 16 + (lane & 15);
  int k = kt * 32 + ((lane >> 4) << 3) + (j & 7) + ((j >> 3) << 4);
  float v = (k < 144) ? w1[n * 144 + k] : 0.0f;
  out[idx] = (_Float16)v;
}

__global__ void prep_w2(const float* __restrict__ w2, _Float16* __restrict__ out) {
  int idx = blockIdx.x * blockDim.x + threadIdx.x;
  if (idx >= 3 * 8 * 512) return;
  int j = idx & 15, lane = (idx >> 4) & 31, frag = idx >> 9;
  int kt = frag % 8, nt = frag / 8;
  int c = nt * 16 + (lane & 15);
  int k = kt * 32 + ((lane >> 4) << 3) + (j & 7) + ((j >> 3) << 4);
  out[idx] = (_Float16)w2[c * 256 + k];
}

// ---- fused NCA step: perception + GEMM1(+bias,leaky) + GEMM2 + mask + residual ----
__global__ __launch_bounds__(TPB) void nca_step(
    const float* __restrict__ xin, float* __restrict__ xout,
    const float* __restrict__ f0, const float* __restrict__ f1,
    const _Float16* __restrict__ w1h, const float* __restrict__ b1,
    const _Float16* __restrict__ w2h, int step)
{
  // xs (38,808B, used through perception) aliases hbuf (33,792B, used from GEMM1)
  __shared__ alignas(16) unsigned char smem[60320];
  float*     xs   = (float*)smem;                    // [3][66][XSTR]
  _Float16*  hbuf = (_Float16*)smem;                 // [64][HSTR]
  _Float16*  pbuf = (_Float16*)(smem + 38816);       // [64][PSTR]

  const int bid = blockIdx.x;
  const int wq = bid & 3;                 // 4 tiles per image row
  const int y  = (bid >> 2) & 255;
  const int b  = bid >> 10;
  const int w0 = wq * PIX;
  const int tid = threadIdx.x;

  // halo tile (reflect padding) -> LDS, channel stride padded to 49 floats
  for (int i = tid; i < 3 * XCOLS * CCH; i += TPB) {
    int ch  = i % CCH;
    int col = (i / CCH) % XCOLS;
    int r   = i / (CCH * XCOLS);
    int gy = y - 1 + r;   gy = gy < 0 ? -gy : (gy > 255 ? 510 - gy : gy);
    int gx = w0 - 1 + col; gx = gx < 0 ? -gx : (gx > 255 ? 510 - gx : gx);
    xs[(r * XCOLS + col) * XSTR + ch] = xin[(((size_t)b * 256 + gy) * 256 + gx) * CCH + ch];
  }
  __syncthreads();

  // perception: each thread owns one pixel; both filters share the 9 taps.
  // filter-weight indices are wave-uniform -> scalar loads.
  {
    const int pix = tid;                  // TPB == PIX
    _Float16* prow = &pbuf[pix * PSTR];
    for (int ch = 0; ch < CCH; ++ch) {
      float t00 = xs[(0 * XCOLS + pix + 0) * XSTR + ch];
      float t01 = xs[(0 * XCOLS + pix + 1) * XSTR + ch];
      float t02 = xs[(0 * XCOLS + pix + 2) * XSTR + ch];
      float t10 = xs[(1 * XCOLS + pix + 0) * XSTR + ch];
      float t11 = xs[(1 * XCOLS + pix + 1) * XSTR + ch];
      float t12 = xs[(1 * XCOLS + pix + 2) * XSTR + ch];
      float t20 = xs[(2 * XCOLS + pix + 0) * XSTR + ch];
      float t21 = xs[(2 * XCOLS + pix + 1) * XSTR + ch];
      float t22 = xs[(2 * XCOLS + pix + 2) * XSTR + ch];
      const float* g0 = &f0[ch * 9];
      const float* g1 = &f1[ch * 9];
      float s0 = g0[0]*t00 + g0[1]*t01 + g0[2]*t02
               + g0[3]*t10 + g0[4]*t11 + g0[5]*t12
               + g0[6]*t20 + g0[7]*t21 + g0[8]*t22;
      float s1 = g1[0]*t00 + g1[1]*t01 + g1[2]*t02
               + g1[3]*t10 + g1[4]*t11 + g1[5]*t12
               + g1[6]*t20 + g1[7]*t21 + g1[8]*t22;
      prow[ch]           = (_Float16)t11;   // identity channels
      prow[CCH + ch]     = (_Float16)s0;    // dwconv0
      prow[2 * CCH + ch] = (_Float16)s1;    // dwconv1
    }
#pragma unroll
    for (int j = 0; j < 8; ++j)             // zero-pad K 144..159
      *(uint32_t*)&prow[144 + 2 * j] = 0u;
  }
  __syncthreads();

  const int lane = tid & 31;
  const int wv   = tid >> 5;
  const int nn   = lane & 15;             // A row / D column / bias index
  const int hi   = (lane & 16) ? 8 : 0;   // A K-offset and D M-offset
  const int pixbase = wv * 32;            // 32 pixels (2 M-tiles) per wave
  const _Float16* arow0 = &pbuf[(pixbase + nn) * PSTR];
  const _Float16* arow1 = &pbuf[(pixbase + 16 + nn) * PSTR];

  // hoist the 2x5 A fragments (shared across all 16 N tiles)
  v16h A0[5], A1[5];
#pragma unroll
  for (int kt = 0; kt < 5; ++kt) {
    A0[kt] = load_afrag(arow0, kt * 32, hi);
    A1[kt] = load_afrag(arow1, kt * 32, hi);
  }

  // GEMM1: h[32x256] = p[32x160] * w1^T ; each B frag feeds two WMMAs
  for (int nt = 0; nt < 16; ++nt) {
    v8f acc0 = {}, acc1 = {};
    const float4* q = (const float4*)(w1h + (size_t)(nt * 5) * 512 + lane * 16);
#pragma unroll
    for (int kt = 0; kt < 5; ++kt) {
      Frag16 bu;
      bu.f4[0] = q[kt * 32 + 0];
      bu.f4[1] = q[kt * 32 + 1];
      acc0 = __builtin_amdgcn_wmma_f32_16x16x32_f16(false, A0[kt], false, bu.v,
                                                    (short)0, acc0, false, false);
      acc1 = __builtin_amdgcn_wmma_f32_16x16x32_f16(false, A1[kt], false, bu.v,
                                                    (short)0, acc1, false, false);
    }
    float bias = b1[nt * 16 + nn];
#pragma unroll
    for (int r = 0; r < 8; ++r) {
      float y0 = acc0[r] + bias;  y0 = y0 > 0.0f ? y0 : 0.01f * y0;
      float y1 = acc1[r] + bias;  y1 = y1 > 0.0f ? y1 : 0.01f * y1;
      hbuf[(pixbase + r + hi) * HSTR + nt * 16 + nn]      = (_Float16)y0;
      hbuf[(pixbase + 16 + r + hi) * HSTR + nt * 16 + nn] = (_Float16)y1;
    }
  }
  __syncthreads();

  // GEMM2: d[32x48] = h[32x256] * w2^T, fire/channel mask, residual add
  const _Float16* hrow0 = &hbuf[(pixbase + nn) * HSTR];
  const _Float16* hrow1 = &hbuf[(pixbase + 16 + nn) * HSTR];
  const uint32_t stepmix = 0x9E3779B9u * (uint32_t)(step + 1);

  for (int nt2 = 0; nt2 < 3; ++nt2) {
    v8f acc0 = {}, acc1 = {};
    const float4* q = (const float4*)(w2h + (size_t)(nt2 * 8) * 512 + lane * 16);
#pragma unroll
    for (int kt = 0; kt < 8; ++kt) {
      v16h a0 = load_afrag(hrow0, kt * 32, hi);
      v16h a1 = load_afrag(hrow1, kt * 32, hi);
      Frag16 bu;
      bu.f4[0] = q[kt * 32 + 0];
      bu.f4[1] = q[kt * 32 + 1];
      acc0 = __builtin_amdgcn_wmma_f32_16x16x32_f16(false, a0, false, bu.v,
                                                    (short)0, acc0, false, false);
      acc1 = __builtin_amdgcn_wmma_f32_16x16x32_f16(false, a1, false, bu.v,
                                                    (short)0, acc1, false, false);
    }
    int c = nt2 * 16 + nn;
#pragma unroll
    for (int r = 0; r < 8; ++r) {
#pragma unroll
      for (int t = 0; t < 2; ++t) {
        int lc = pixbase + t * 16 + r + hi;           // local pixel in block tile
        uint32_t pixg = ((uint32_t)b * 256 + y) * 256 + (w0 + lc);
        float d = (t == 0) ? acc0[r] : acc1[r];
        if (c < 3) d = 0.0f;                          // immutable image channels
        float fire = (hashu(pixg + stepmix) >> 31) ? 1.0f : 0.0f;  // uniform > 0.5
        float xv = xin[(size_t)pixg * CCH + c];       // L2-resident re-read
        xout[(size_t)pixg * CCH + c] = xv + d * fire;
      }
    }
  }
}

extern "C" void kernel_launch(void* const* d_in, const int* in_sizes, int n_in,
                              void* d_out, int out_size, void* d_ws, size_t ws_size,
                              hipStream_t stream) {
  const float* x  = (const float*)d_in[0];
  const float* f0 = (const float*)d_in[1];
  const float* f1 = (const float*)d_in[2];
  const float* w1 = (const float*)d_in[3];
  const float* b1 = (const float*)d_in[4];
  const float* w2 = (const float*)d_in[5];
  float* out = (float*)d_out;

  char* ws = (char*)d_ws;
  const size_t XB = (size_t)8 * 256 * 256 * CCH * sizeof(float);  // 100,663,296 B
  float*     xbuf = (float*)ws;
  _Float16*  w1h  = (_Float16*)(ws + XB);
  _Float16*  w2h  = (_Float16*)(ws + XB + (size_t)16 * 5 * 512 * 2);

  prep_w1<<<160, 256, 0, stream>>>(w1, w1h);
  prep_w2<<<48, 256, 0, stream>>>(w2, w2h);

  dim3 grid(8 * 256 * (256 / PIX));   // 8192 blocks
  dim3 block(TPB);
  // steps == 4 per setup_inputs; ping-pong x between workspace and d_out
  nca_step<<<grid, block, 0, stream>>>(x,    xbuf, f0, f1, w1h, b1, w2h, 0);
  nca_step<<<grid, block, 0, stream>>>(xbuf, out,  f0, f1, w1h, b1, w2h, 1);
  nca_step<<<grid, block, 0, stream>>>(out,  xbuf, f0, f1, w1h, b1, w2h, 2);
  nca_step<<<grid, block, 0, stream>>>(xbuf, out,  f0, f1, w1h, b1, w2h, 3);
}
